// TinyARPolicy_31112743092608
// MI455X (gfx1250) — matrix operand order repaired and checked by reading, not compiled, not run on hardware
//
#include <hip/hip_runtime.h>
#include <hip/hip_bf16.h>

// ---------------------------------------------------------------------------
// TinyARPolicy on MI455X (gfx1250): persistent per-wave LSTM sampler.
// Each wave32 owns 32 batch rows (two 16-row WMMA M-tiles) so every LDS
// weight-fragment load feeds two v_wmma_f32_16x16x32_f16 ops. Gates computed
// in two passes (i,g then f,o) to cap live accumulators at 32 VGPRs.
// Nonlinearities use hardware transcendentals (v_exp/v_rcp/v_tanh) instead
// of libm division sequences.
// ---------------------------------------------------------------------------

typedef _Float16 v16h __attribute__((ext_vector_type(16)));
typedef _Float16 v8h  __attribute__((ext_vector_type(8)));
typedef float    v8f  __attribute__((ext_vector_type(8)));

#define BLOCK 256
#define NW 8                      // waves per block
static constexpr int VOCAB = 32;
static constexpr int HID   = 64;
static constexpr int LEN   = 16;
static constexpr int BOS   = 32;  // embed row index for BOS
static constexpr int MROWS = 32;  // batch rows per wave (2 WMMA M-tiles)

// ---- LDS layout (bytes), all offsets 16B aligned --------------------------
// Weight B-fragments: frag = (nt, kf), 32 lanes x 16 halves (32B/lane) = 1KB.
// wih/whh: 16 nt x 2 kf = 32 frags = 32KB each.  head: 2 nt x 2 kf = 4KB.
static constexpr int OFF_WIH  = 0;
static constexpr int OFF_WHH  = OFF_WIH + 32 * 1024;        // 32768
static constexpr int OFF_HW   = OFF_WHH + 32 * 1024;        // 65536
static constexpr int OFF_EMB  = OFF_HW  + 4 * 1024;         // 69632 (33*64 f16 = 4224)
static constexpr int OFF_BIAS = OFF_EMB + 4224;             // 73856 (256 f32)
static constexpr int OFF_HB   = OFF_BIAS + 1024;            // 74880 (32 f32)
static constexpr int OFF_WAVE = OFF_HB + 128;               // 75008
static constexpr int PW_HBUF  = MROWS * 64 * 2;             // 4096: h f16 [32][64]
static constexpr int PW_LBUF  = MROWS * 32 * 4;             // 4096: logits f32 [32][32]
static constexpr int PW_TOK   = 128;                        // 32 ints
static constexpr int PW       = PW_HBUF + PW_LBUF + PW_TOK; // 8320
static constexpr int LDS_BYTES = OFF_WAVE + NW * PW;        // 141568

__device__ __forceinline__ v16h cat16(v8h a, v8h b) {
  return __builtin_shufflevector(a, b, 0,1,2,3,4,5,6,7,8,9,10,11,12,13,14,15);
}

// sigmoid via v_exp_f32 + v_rcp_f32 (no IEEE div-fixup chain)
__device__ __forceinline__ float sigf(float x) {
  return __builtin_amdgcn_rcpf(1.0f + __expf(-x));
}
// tanh via hardware v_tanh_f32 when the builtin exists; div-free fallback.
__device__ __forceinline__ float tanh_fast(float x) {
#if __has_builtin(__builtin_amdgcn_tanhf)
  return __builtin_amdgcn_tanhf(x);
#else
  return 1.0f - 2.0f * __builtin_amdgcn_rcpf(__expf(2.0f * x) + 1.0f);
#endif
}

__global__ void __launch_bounds__(BLOCK)
tiny_ar_policy_kernel(const float* __restrict__ embed,
                      const float* __restrict__ w_ih,
                      const float* __restrict__ w_hh,
                      const float* __restrict__ b_ih,
                      const float* __restrict__ b_hh,
                      const float* __restrict__ head_w,
                      const float* __restrict__ head_b,
                      const float* __restrict__ gum,
                      int*   __restrict__ toks,
                      float* __restrict__ logp,
                      int batchN)
{
  extern __shared__ char smem[];
  _Float16* wihB  = (_Float16*)(smem + OFF_WIH);
  _Float16* whhB  = (_Float16*)(smem + OFF_WHH);
  _Float16* headB = (_Float16*)(smem + OFF_HW);
  _Float16* embH  = (_Float16*)(smem + OFF_EMB);
  float*    biasS = (float*)(smem + OFF_BIAS);
  float*    hbS   = (float*)(smem + OFF_HB);

  const int tid = threadIdx.x;

  // ---- one-time f32 -> f16 LDS staging (swizzled into B-fragment layout) --
  // Fragment element j: nt=j>>10, kf=(j>>9)&1, lane=(j>>4)&31, v=j&15.
  // B[k][n] = W[n][k] with n = nt*16 + (lane&15), k = kf*32 + (lane>>4)*16 + v.
  for (int j = tid; j < 16384; j += BLOCK) {
    int nt = j >> 10, kf = (j >> 9) & 1, ln = (j >> 4) & 31, v = j & 15;
    int n = nt * 16 + (ln & 15);
    int k = kf * 32 + ((ln >> 4) << 4) + v;
    wihB[j] = (_Float16)w_ih[n * 64 + k];
    whhB[j] = (_Float16)w_hh[n * 64 + k];
  }
  for (int j = tid; j < 2048; j += BLOCK) {
    int nt = j >> 10, kf = (j >> 9) & 1, ln = (j >> 4) & 31, v = j & 15;
    int n = nt * 16 + (ln & 15);
    int k = kf * 32 + ((ln >> 4) << 4) + v;
    headB[j] = (_Float16)head_w[n * 64 + k];
  }
  for (int j = tid; j < 33 * 64; j += BLOCK) embH[j] = (_Float16)embed[j];
  for (int j = tid; j < 256; j += BLOCK)     biasS[j] = b_ih[j] + b_hh[j];
  for (int j = tid; j < 32; j += BLOCK)      hbS[j]   = head_b[j];
  __syncthreads();

  // ---- per-wave setup ------------------------------------------------------
  const int w    = tid >> 5;
  const int lane = tid & 31;
  const int waveRow0 = (blockIdx.x * NW + w) * MROWS;
  _Float16* hbuf = (_Float16*)(smem + OFF_WAVE + w * PW);
  float*    lbuf = (float*)(smem + OFF_WAVE + w * PW + PW_HBUF);
  int*      tokb = (int*)(smem + OFF_WAVE + w * PW + PW_HBUF + PW_LBUF);

  tokb[lane] = BOS;

  const int myrow = lane & 15;
  const int hi    = lane >> 4;     // 0: lanes 0-15, 1: lanes 16-31
  const int rbase = hi * 8;        // D-layout: VGPR r holds row r + 8*hi

  // Load WMMA B fragment (nt, kf) from a swizzled LDS weight region: one
  // contiguous 32B chunk per lane -> 2x ds_load_b128.
  auto loadB = [&](const _Float16* base, int nt, int kf) -> v16h {
    const _Float16* p = base + (nt * 2 + kf) * 512 + lane * 16;
    return cat16(*(const v8h*)p, *(const v8h*)(p + 8));
  };
  // Load WMMA A fragment kf from a row-major [.][64] f16 buffer row.
  // Contiguous k order used here (same as the B packing): lane<16 holds K kb+0..15, lane>=16 kb+16..31.
  auto loadA = [&](const _Float16* rowbase, int kf) -> v16h {
    int o1 = kf * 32 + (hi << 4);
    return cat16(*(const v8h*)(rowbase + o1), *(const v8h*)(rowbase + o1 + 8));
  };

  // c state for two M-tiles, D layout, cols cg*16+(lane&15)
  v8f CstA[4], CstB[4];
  #pragma unroll
  for (int g = 0; g < 4; ++g)
    #pragma unroll
    for (int r = 0; r < 8; ++r) { CstA[g][r] = 0.0f; CstB[g][r] = 0.0f; }

  // h A-fragments for both tiles (h0 = 0)
  v16h HfA0, HfA1, HfB0, HfB1;
  #pragma unroll
  for (int i = 0; i < 16; ++i) {
    HfA0[i] = (_Float16)0.0f; HfA1[i] = (_Float16)0.0f;
    HfB0[i] = (_Float16)0.0f; HfB1[i] = (_Float16)0.0f;
  }

  float logpAcc = 0.0f;

  for (int t = 0; t < LEN; ++t) {
    // Prefetch this step's Gumbel rows (consumed after the matmuls).
    {
      const float* grow = gum + ((size_t)t * batchN + (waveRow0 + lane)) * VOCAB;
      __builtin_prefetch(grow, 0, 0);
    }

    // ---- embedding A-fragments for current tokens (both tiles) ------------
    int tokA = tokb[myrow];
    int tokB = tokb[16 + myrow];
    const _Float16* erowA = embH + tokA * 64;
    const _Float16* erowB = embH + tokB * 64;
    v16h EfA0 = loadA(erowA, 0), EfA1 = loadA(erowA, 1);
    v16h EfB0 = loadA(erowB, 0), EfB1 = loadA(erowB, 1);

    // One gate tile for both M-tiles; each B fragment feeds 2 WMMAs.
    auto gacc2 = [&](int nt, v8f& accA, v8f& accB) {
      float bv = biasS[nt * 16 + myrow];
      #pragma unroll
      for (int r = 0; r < 8; ++r) { accA[r] = bv; accB[r] = bv; }
      v16h b0 = loadB(wihB, nt, 0);
      accA = __builtin_amdgcn_wmma_f32_16x16x32_f16(false, EfA0, false, b0, (short)0, accA, false, false);
      accB = __builtin_amdgcn_wmma_f32_16x16x32_f16(false, EfB0, false, b0, (short)0, accB, false, false);
      v16h b1 = loadB(wihB, nt, 1);
      accA = __builtin_amdgcn_wmma_f32_16x16x32_f16(false, EfA1, false, b1, (short)0, accA, false, false);
      accB = __builtin_amdgcn_wmma_f32_16x16x32_f16(false, EfB1, false, b1, (short)0, accB, false, false);
      v16h b2 = loadB(whhB, nt, 0);
      accA = __builtin_amdgcn_wmma_f32_16x16x32_f16(false, HfA0, false, b2, (short)0, accA, false, false);
      accB = __builtin_amdgcn_wmma_f32_16x16x32_f16(false, HfB0, false, b2, (short)0, accB, false, false);
      v16h b3 = loadB(whhB, nt, 1);
      accA = __builtin_amdgcn_wmma_f32_16x16x32_f16(false, HfA1, false, b3, (short)0, accA, false, false);
      accB = __builtin_amdgcn_wmma_f32_16x16x32_f16(false, HfB1, false, b3, (short)0, accB, false, false);
    };

    // ---- LSTM: 16-col groups; two passes to limit live accumulators -------
    #pragma unroll
    for (int cg = 0; cg < 4; ++cg) {
      // pass 1: input gate (nt=cg) and cell gate (nt=cg+8)
      float p1A[8], p1B[8];
      {
        v8f iA, iB, gA, gB;
        gacc2(cg,     iA, iB);
        gacc2(cg + 8, gA, gB);
        #pragma unroll
        for (int r = 0; r < 8; ++r) {
          p1A[r] = sigf(iA[r]) * tanh_fast(gA[r]);
          p1B[r] = sigf(iB[r]) * tanh_fast(gB[r]);
        }
      }
      // pass 2: forget gate (nt=cg+4) and output gate (nt=cg+12)
      {
        v8f fA, fB, oA, oB;
        gacc2(cg + 4,  fA, fB);
        gacc2(cg + 12, oA, oB);
        int colh = cg * 16 + myrow;
        #pragma unroll
        for (int r = 0; r < 8; ++r) {
          float cnA = sigf(fA[r]) * CstA[cg][r] + p1A[r];
          float cnB = sigf(fB[r]) * CstB[cg][r] + p1B[r];
          CstA[cg][r] = cnA;
          CstB[cg][r] = cnB;
          float hvA = sigf(oA[r]) * tanh_fast(cnA);
          float hvB = sigf(oB[r]) * tanh_fast(cnB);
          hbuf[(rbase + r) * 64 + colh]        = (_Float16)hvA;
          hbuf[(16 + rbase + r) * 64 + colh]   = (_Float16)hvB;
        }
      }
    }

    // ---- re-read new h as A-fragments (in-wave DS ordering, no barrier) ---
    const _Float16* hrowA = hbuf + myrow * 64;
    const _Float16* hrowB = hbuf + (16 + myrow) * 64;
    HfA0 = loadA(hrowA, 0); HfA1 = loadA(hrowA, 1);
    HfB0 = loadA(hrowB, 0); HfB1 = loadA(hrowB, 1);

    // ---- head: logits[32,32] ----------------------------------------------
    #pragma unroll
    for (int n = 0; n < 2; ++n) {
      float bv = hbS[n * 16 + myrow];
      v8f accA, accB;
      #pragma unroll
      for (int r = 0; r < 8; ++r) { accA[r] = bv; accB[r] = bv; }
      v16h b0 = loadB(headB, n, 0);
      accA = __builtin_amdgcn_wmma_f32_16x16x32_f16(false, HfA0, false, b0, (short)0, accA, false, false);
      accB = __builtin_amdgcn_wmma_f32_16x16x32_f16(false, HfB0, false, b0, (short)0, accB, false, false);
      v16h b1 = loadB(headB, n, 1);
      accA = __builtin_amdgcn_wmma_f32_16x16x32_f16(false, HfA1, false, b1, (short)0, accA, false, false);
      accB = __builtin_amdgcn_wmma_f32_16x16x32_f16(false, HfB1, false, b1, (short)0, accB, false, false);
      #pragma unroll
      for (int r = 0; r < 8; ++r) {
        lbuf[(rbase + r) * 32 + n * 16 + myrow]      = accA[r];
        lbuf[(16 + rbase + r) * 32 + n * 16 + myrow] = accB[r];
      }
    }

    // ---- Gumbel-max sample + log-softmax: one lane owns one full row ------
    {
      int rowG = waveRow0 + lane;
      const float* lrow = lbuf + lane * 32;
      float lg[32];
      #pragma unroll
      for (int q = 0; q < 8; ++q) {
        float4 f = ((const float4*)lrow)[q];
        lg[4*q] = f.x; lg[4*q+1] = f.y; lg[4*q+2] = f.z; lg[4*q+3] = f.w;
      }
      const float* grow = gum + ((size_t)t * batchN + rowG) * VOCAB;
      float lmax = -3.4e38f, smax = -3.4e38f, slog = 0.0f;
      int sidx = 0;
      #pragma unroll
      for (int q = 0; q < 8; ++q) {
        float4 g = ((const float4*)grow)[q];
        float gs[4] = {g.x, g.y, g.z, g.w};
        #pragma unroll
        for (int e = 0; e < 4; ++e) {
          int j = 4 * q + e;
          lmax = fmaxf(lmax, lg[j]);
          float sv = lg[j] + gs[e];
          if (sv > smax) { smax = sv; sidx = j; slog = lg[j]; }
        }
      }
      float se = 0.0f;
      #pragma unroll
      for (int j = 0; j < 32; ++j) se += __expf(lg[j] - lmax);
      logpAcc += slog - lmax - __logf(se);
      tokb[lane] = sidx;
      toks[(size_t)rowG * LEN + t] = sidx;
    }
  }

  logp[waveRow0 + lane] = logpAcc;
}

extern "C" void kernel_launch(void* const* d_in, const int* in_sizes, int n_in,
                              void* d_out, int out_size, void* d_ws, size_t ws_size,
                              hipStream_t stream) {
  const float* embed  = (const float*)d_in[0];
  const float* w_ih   = (const float*)d_in[1];
  const float* w_hh   = (const float*)d_in[2];
  const float* b_ih   = (const float*)d_in[3];
  const float* b_hh   = (const float*)d_in[4];
  const float* head_w = (const float*)d_in[5];
  const float* head_b = (const float*)d_in[6];
  const float* gum    = (const float*)d_in[7];

  int batchN = in_sizes[7] / (LEN * VOCAB);   // gumbel is [L, B, V]

  int*   toks = (int*)d_out;                          // [B, L] int32 tokens
  float* logp = (float*)d_out + (size_t)batchN * LEN; // [B] float32 logp

  (void)hipFuncSetAttribute((const void*)tiny_ar_policy_kernel,
                            hipFuncAttributeMaxDynamicSharedMemorySize,
                            LDS_BYTES);

  dim3 grid(batchN / (NW * MROWS));   // 131072 / 256 = 512 blocks
  tiny_ar_policy_kernel<<<grid, BLOCK, LDS_BYTES, stream>>>(
      embed, w_ih, w_hh, b_ih, b_hh, head_w, head_b, gum, toks, logp, batchN);
}
